// NeuralMemoryLM_6012954214667
// MI455X (gfx1250) — matrix operand order repaired and checked by reading, not compiled, not run on hardware
//
#include <hip/hip_runtime.h>
#include <hip/hip_bf16.h>

// ---------------- model constants (match reference) ----------------
#define DEPTH_  4
#define DIM_    512
#define HID_    2048
#define VOCAB_  32000
#define CHUNK_  64
#define BB_     2
#define LL_     2048
#define NC_     (LL_ / CHUNK_)     // 32 chunks
#define BL_     (BB_ * LL_)        // 4096 rows
#define EPS_    1e-6f

typedef __bf16 bf16;
typedef __attribute__((ext_vector_type(16))) __bf16 v16bf;
typedef __attribute__((ext_vector_type(8)))  float  v8f;
typedef __attribute__((ext_vector_type(4)))  unsigned int v4u;

// ---------------- small helpers ----------------
__device__ inline float wave_sum(float v) {
  #pragma unroll
  for (int o = 16; o > 0; o >>= 1) v += __shfl_xor(v, o, 32);
  return v;
}

// ---------------- WMMA fragment loads (bf16, K-contiguous rows) ----------------
// A: 16x32 tile of A[M,K] row-major.  lanes 0-15: row m, K 0..7 / 16..23
//                                     lanes16-31: row m, K 8..15 / 24..31
__device__ inline v16bf load_a_frag(const bf16* __restrict__ A, int lda,
                                    int row0, int k0, int lane, int M) {
  int half = lane >> 4;
  int m = row0 + (lane & 15);
  if (m >= M) m = M - 1;
  const bf16* p = A + (size_t)m * lda + k0 + half * 8;
  union { v16bf v; struct { v4u lo, hi; } q; } f;
  f.q.lo = *(const v4u*)p;
  f.q.hi = *(const v4u*)(p + 16);
  return f.v;
}
// B: 32x16 tile of W[N,K]^T.  lane n (0-15): column n = W row n, K 0..7 / 16..23;
//                             lane n+16:     same column, K 8..15 / 24..31
__device__ inline v16bf load_b_frag(const bf16* __restrict__ W, int ldb,
                                    int col0, int k0, int lane, int N) {
  int half = lane >> 4;
  int n = col0 + (lane & 15);
  if (n >= N) n = N - 1;
  const bf16* p = W + (size_t)n * ldb + k0 + half * 8;
  union { v16bf v; struct { v4u lo, hi; } q; } f;
  f.q.lo = *(const v4u*)p;
  f.q.hi = *(const v4u*)(p + 16);
  return f.v;
}

// ---------------- generic batched GEMM: C[M,N] = A[M,K] @ W[N,K]^T (+epilogue) ---
// 8 waves: 2 row-groups x 4 col-groups. Wave tile = 32 x (16*JT).
// Block tile = 64 x (64*JT).  JT=4 for large-N GEMMs, JT=2 for the tiny kq GEMM.
// K-loop is explicitly 2x unrolled (64 K / iter): both half-steps' fragments are
// issued before the WMMA chains so half-2 loads overlap half-1 matrix ops.
// K must be a multiple of 64 (all K here are 64 / 512 / 2048).
// EPI 0: C = acc                 (OutT = float or bf16)
// EPI 1: C = acc + bias[n]
// EPI 2: C += acc + bias[n]      (fused residual accumulate)
// EPI 3: C = C*scal[z] + acc     (carry-W decay update, in-place, bf16)
template <int EPI, typename OutT, int JT>
__global__ void __launch_bounds__(256)
gemm_wmma(const bf16* __restrict__ A, const bf16* __restrict__ Bw,
          OutT* __restrict__ C, int M, int N, int K,
          int lda, int ldb, int ldc,
          long long sA, long long sB, long long sC,
          const float* __restrict__ bias,
          const float* __restrict__ scal, int scalStride) {
  const bf16* Ab = A  + (long long)blockIdx.z * sA;
  const bf16* Bb = Bw + (long long)blockIdx.z * sB;
  OutT*       Cb = C  + (long long)blockIdx.z * sC;
  int lane = threadIdx.x & 31;
  int wave = threadIdx.x >> 5;
  int rowBase = blockIdx.x * 64 + (wave & 1) * 32;
  int colBase = blockIdx.y * (64 * JT) + (wave >> 1) * (16 * JT);

  v8f acc[2][JT];
  #pragma unroll
  for (int i = 0; i < 2; i++)
    #pragma unroll
    for (int j = 0; j < JT; j++)
      #pragma unroll
      for (int r = 0; r < 8; r++) acc[i][j][r] = 0.f;

  for (int k0 = 0; k0 < K; k0 += 64) {
    if (k0 + 64 < K) {
      int pm = rowBase + (lane & 15); if (pm >= M) pm = M - 1;
      int pn = colBase + (lane & 15); if (pn >= N) pn = N - 1;
      __builtin_prefetch(Ab + (size_t)pm * lda + k0 + 64, 0, 3);
      __builtin_prefetch(Bb + (size_t)pn * ldb + k0 + 64, 0, 3);
    }
    // ---- issue all fragment loads for BOTH half-steps ----
    v16bf a0 = load_a_frag(Ab, lda, rowBase,      k0,      lane, M);
    v16bf a1 = load_a_frag(Ab, lda, rowBase + 16, k0,      lane, M);
    v16bf a2 = load_a_frag(Ab, lda, rowBase,      k0 + 32, lane, M);
    v16bf a3 = load_a_frag(Ab, lda, rowBase + 16, k0 + 32, lane, M);
    v16bf b0[JT], b1[JT];
    #pragma unroll
    for (int j = 0; j < JT; j++)
      b0[j] = load_b_frag(Bb, ldb, colBase + j * 16, k0,      lane, N);
    #pragma unroll
    for (int j = 0; j < JT; j++)
      b1[j] = load_b_frag(Bb, ldb, colBase + j * 16, k0 + 32, lane, N);
    // ---- half-step 1 WMMA chain (half-step 2 loads still in flight) ----
    #pragma unroll
    for (int j = 0; j < JT; j++) {
      acc[0][j] = __builtin_amdgcn_wmma_f32_16x16x32_bf16(false, a0, false, b0[j], (short)0, acc[0][j], false, false);
      acc[1][j] = __builtin_amdgcn_wmma_f32_16x16x32_bf16(false, a1, false, b0[j], (short)0, acc[1][j], false, false);
    }
    // ---- half-step 2 WMMA chain ----
    #pragma unroll
    for (int j = 0; j < JT; j++) {
      acc[0][j] = __builtin_amdgcn_wmma_f32_16x16x32_bf16(false, a2, false, b1[j], (short)0, acc[0][j], false, false);
      acc[1][j] = __builtin_amdgcn_wmma_f32_16x16x32_bf16(false, a3, false, b1[j], (short)0, acc[1][j], false, false);
    }
  }

  int half = lane >> 4, nn = lane & 15;
  float sc = (EPI == 3) ? scal[(long long)blockIdx.z * scalStride] : 0.f;
  #pragma unroll
  for (int i = 0; i < 2; i++)
    #pragma unroll
    for (int j = 0; j < JT; j++) {
      int col = colBase + j * 16 + nn;
      #pragma unroll
      for (int r = 0; r < 8; r++) {
        int row = rowBase + i * 16 + half * 8 + r;   // C layout: VGPR r -> M=r / r+8
        if (row < M && col < N) {
          size_t idx = (size_t)row * ldc + col;
          float v = acc[i][j][r];
          if (EPI == 0)      Cb[idx] = (OutT)v;
          else if (EPI == 1) Cb[idx] = (OutT)(v + bias[col]);
          else if (EPI == 2) Cb[idx] = (OutT)((float)Cb[idx] + v + bias[col]);
          else               Cb[idx] = (OutT)((float)Cb[idx] * sc + v);
        }
      }
    }
}

// ---------------- elementwise / reduction kernels ----------------
__global__ void f2bf_kernel(const float* __restrict__ in, bf16* __restrict__ out, long long n) {
  long long i = (long long)blockIdx.x * blockDim.x + threadIdx.x;
  if (i < n) out[i] = (bf16)in[i];
}

// embedding gather with max_norm=1 renorm; one block per token
__global__ void embed_kernel(const int* __restrict__ tok, const float* __restrict__ tab,
                             float* __restrict__ x) {
  int row = blockIdx.x;
  const float* e = tab + (size_t)tok[row] * DIM_;
  float ss = 0.f;
  for (int i = threadIdx.x; i < DIM_; i += blockDim.x) { float v = e[i]; ss += v * v; }
  ss = wave_sum(ss);
  __shared__ float red[8];
  __shared__ float scale_s;
  int wid = threadIdx.x >> 5, lane = threadIdx.x & 31;
  if (lane == 0) red[wid] = ss;
  __syncthreads();
  if (threadIdx.x == 0) {
    float t = 0.f;
    #pragma unroll
    for (int w = 0; w < 8; w++) t += red[w];
    float nrm = sqrtf(t);
    scale_s = (nrm > 1.f) ? 1.f / (nrm + 1e-7f) : 1.f;
  }
  __syncthreads();
  float sc = scale_s;
  float* xr = x + (size_t)row * DIM_;
  for (int i = threadIdx.x; i < DIM_; i += blockDim.x) xr[i] = e[i] * sc;
}

// RMS norm row of DIM_, fp32 in -> bf16 out; one block per row
__global__ void rms_kernel(const float* __restrict__ x, const float* __restrict__ w,
                           bf16* __restrict__ out) {
  int row = blockIdx.x;
  const float* xr = x + (size_t)row * DIM_;
  float ss = 0.f;
  for (int i = threadIdx.x; i < DIM_; i += blockDim.x) { float v = xr[i]; ss += v * v; }
  ss = wave_sum(ss);
  __shared__ float red[8];
  __shared__ float scale_s;
  int wid = threadIdx.x >> 5, lane = threadIdx.x & 31;
  if (lane == 0) red[wid] = ss;
  __syncthreads();
  if (threadIdx.x == 0) {
    float t = 0.f;
    #pragma unroll
    for (int w8 = 0; w8 < 8; w8++) t += red[w8];
    scale_s = rsqrtf(t / (float)DIM_ + EPS_);
  }
  __syncthreads();
  float sc = scale_s;
  bf16* o = out + (size_t)row * DIM_;
  for (int i = threadIdx.x; i < DIM_; i += blockDim.x) o[i] = (bf16)(xr[i] * sc * w[i]);
}

// lr / log_wd GEMV: one wave per (b,l) row
__global__ void lrwd_kernel(const bf16* __restrict__ xn, const float* __restrict__ lrw,
                            const float* __restrict__ wdw, const float* __restrict__ lrb,
                            const float* __restrict__ wdb, const float* __restrict__ lbl,
                            const float* __restrict__ lbw,
                            float* __restrict__ lr, float* __restrict__ logwd) {
  int r = blockIdx.x * (blockDim.x >> 5) + (threadIdx.x >> 5);
  int lane = threadIdx.x & 31;
  if (r >= BL_) return;
  const bf16* xr = xn + (size_t)r * DIM_;
  float s1 = 0.f, s2 = 0.f;
  for (int i = lane; i < DIM_; i += 32) {
    float v = (float)xr[i];
    s1 += v * lrw[i];
    s2 += v * wdw[i];
  }
  s1 = wave_sum(s1); s2 = wave_sum(s2);
  if (lane == 0) {
    float z1 = s1 + lrb[0];
    lr[r] = expf(lbl[0]) / (1.f + expf(-z1));
    float z2 = s2 + wdb[0];
    float ls = (z2 >= 0.f) ? -log1pf(expf(-z2)) : (z2 - log1pf(expf(z2)));
    logwd[r] = lbw[0] + ls;
  }
}

// per-chunk cumulative decay: wd_cross = exp(cumsum(log_wd)) within each chunk
__global__ void wdscan_kernel(const float* __restrict__ logwd,
                              float* __restrict__ wdcross, float* __restrict__ wdlast) {
  int id = blockIdx.x * blockDim.x + threadIdx.x;
  if (id >= BB_ * NC_) return;
  int b = id / NC_, c = id % NC_;
  const float* lw = logwd + (size_t)b * LL_ + c * CHUNK_;
  float* wc = wdcross + (size_t)b * LL_ + c * CHUNK_;
  float acc = 0.f;
  for (int i = 0; i < CHUNK_; i++) { acc += lw[i]; wc[i] = expf(acc); }
  wdlast[b * NC_ + c] = expf(acc);
}

// k[B,L,H] bf16 -> kT[B,NC,H,CHUNK] bf16
__global__ void ktrans_kernel(const bf16* __restrict__ k, bf16* __restrict__ kT) {
  long long i = (long long)blockIdx.x * blockDim.x + threadIdx.x;
  const long long total = (long long)BB_ * LL_ * HID_;
  if (i >= total) return;
  int l = (int)(i % CHUNK_);
  int h = (int)((i / CHUNK_) % HID_);
  int c = (int)((i / ((long long)CHUNK_ * HID_)) % NC_);
  int b = (int)(i / ((long long)CHUNK_ * HID_ * NC_));
  kT[i] = k[((size_t)b * LL_ + (size_t)c * CHUNK_ + l) * HID_ + h];
}

// msum[b,l] = sum_{m>=l} kq[b,l,m] * wd_cross[m]/wd_cross[l]; one wave per row
__global__ void msum_kernel(const float* __restrict__ kq, const float* __restrict__ wdc,
                            float* __restrict__ msum, int chunkoff) {
  int r = blockIdx.x * (blockDim.x >> 5) + (threadIdx.x >> 5);
  int lane = threadIdx.x & 31;
  if (r >= BB_ * CHUNK_) return;
  int b = r / CHUNK_, l = r % CHUNK_;
  const float* kqr = kq + ((size_t)b * CHUNK_ + l) * CHUNK_;
  const float* wc = wdc + (size_t)b * LL_ + chunkoff;
  float s = 0.f;
  for (int m = lane; m < CHUNK_; m += 32)
    if (m >= l) s += kqr[m] * wc[m];
  s = wave_sum(s);
  if (lane == 0) msum[r] = s / wc[l];
}

// y = WpK*(wd_cross - lr*msum) + v*lr*msum; residual-add into x.
// Also emit AscT[d,l] = -(WpK - v)*lr*wd_last/wd_cross  (bf16, transposed for W-update GEMM)
__global__ void yepi_kernel(const float* __restrict__ WpK, const float* __restrict__ vf,
                            const float* __restrict__ lr, const float* __restrict__ wdc,
                            const float* __restrict__ msum, const float* __restrict__ wdlast,
                            float* __restrict__ x, bf16* __restrict__ AscT,
                            int chunkoff, int cidx) {
  int i = blockIdx.x * blockDim.x + threadIdx.x;
  if (i >= BB_ * CHUNK_ * DIM_) return;
  int d = i % DIM_;
  int l = (i / DIM_) % CHUNK_;
  int b = i / (DIM_ * CHUNK_);
  size_t gi = (size_t)b * LL_ + chunkoff + l;
  float wpk = WpK[((size_t)b * CHUNK_ + l) * DIM_ + d];
  float vv  = vf[gi * DIM_ + d];
  float lrl = lr[gi];
  float wcl = wdc[gi];
  float ms  = msum[b * CHUNK_ + l];
  float wl  = wdlast[b * NC_ + cidx];
  x[gi * DIM_ + d] += wpk * (wcl - lrl * ms) + vv * lrl * ms;
  AscT[((size_t)b * DIM_ + d) * CHUNK_ + l] = (bf16)(-(wpk - vv) * lrl * wl / wcl);
}

// W-carry init: broadcast fp32 W_init to both batch slots in bf16
__global__ void wcinit_kernel(const float* __restrict__ Wi, bf16* __restrict__ Wc) {
  int i = blockIdx.x * blockDim.x + threadIdx.x;
  if (i < DIM_ * HID_) {
    bf16 v = (bf16)Wi[i];
    Wc[i] = v;
    Wc[i + DIM_ * HID_] = v;
  }
}

// FFN gate: g = h1 * silu(ha) -> bf16
__global__ void gate_kernel(const float* __restrict__ h1, const float* __restrict__ ha,
                            bf16* __restrict__ g, long long n) {
  long long i = (long long)blockIdx.x * blockDim.x + threadIdx.x;
  if (i < n) {
    float a = ha[i];
    g[i] = (bf16)(h1[i] * a / (1.f + expf(-a)));
  }
}

// ---------------- host orchestration ----------------
extern "C" void kernel_launch(void* const* d_in, const int* in_sizes, int n_in,
                              void* d_out, int out_size, void* d_ws, size_t ws_size,
                              hipStream_t stream) {
  (void)in_sizes; (void)n_in; (void)out_size; (void)ws_size;

  const int*   tokens   = (const int*)d_in[0];
  const float* emb      = (const float*)d_in[1];
  const float* Wq       = (const float*)d_in[2];
  const float* Wk       = (const float*)d_in[3];
  const float* Wv       = (const float*)d_in[4];
  const float* lr_w     = (const float*)d_in[5];
  const float* lr_b     = (const float*)d_in[6];
  const float* wd_w     = (const float*)d_in[7];
  const float* wd_b     = (const float*)d_in[8];
  const float* lbl      = (const float*)d_in[9];
  const float* lbw      = (const float*)d_in[10];
  const float* W_init   = (const float*)d_in[11];
  const float* norm_mem = (const float*)d_in[12];
  const float* norm_ffn = (const float*)d_in[13];
  const float* ffn_w1   = (const float*)d_in[14];
  const float* ffn_b1   = (const float*)d_in[15];
  const float* ffn_wa   = (const float*)d_in[16];
  const float* ffn_ba   = (const float*)d_in[17];
  const float* ffn_wo   = (const float*)d_in[18];
  const float* ffn_bo   = (const float*)d_in[19];
  const float* norm_last= (const float*)d_in[20];
  const float* out_w    = (const float*)d_in[21];
  const float* out_b    = (const float*)d_in[22];
  float* logits = (float*)d_out;

  // ---- carve workspace ----
  char* wp = (char*)d_ws;
  auto carve = [&](size_t elems, size_t esz) -> void* {
    void* r = (void*)wp;
    wp += (elems * esz + 255) & ~(size_t)255;
    return r;
  };
  bf16*  bWq  = (bf16*)carve((size_t)DEPTH_ * HID_ * DIM_, 2);
  bf16*  bWk  = (bf16*)carve((size_t)DEPTH_ * HID_ * DIM_, 2);
  bf16*  bWv  = (bf16*)carve((size_t)DEPTH_ * DIM_ * DIM_, 2);
  bf16*  bW1  = (bf16*)carve((size_t)DEPTH_ * HID_ * DIM_, 2);
  bf16*  bWa  = (bf16*)carve((size_t)DEPTH_ * HID_ * DIM_, 2);
  bf16*  bWo  = (bf16*)carve((size_t)DEPTH_ * DIM_ * HID_, 2);
  bf16*  bOw  = (bf16*)carve((size_t)VOCAB_ * DIM_, 2);
  float* x    = (float*)carve((size_t)BL_ * DIM_, 4);
  bf16*  xn   = (bf16*) carve((size_t)BL_ * DIM_, 2);
  bf16*  qb   = (bf16*) carve((size_t)BL_ * HID_, 2);
  bf16*  kb   = (bf16*) carve((size_t)BL_ * HID_, 2);
  bf16*  kT   = (bf16*) carve((size_t)BL_ * HID_, 2);
  float* vf   = (float*)carve((size_t)BL_ * DIM_, 4);
  float* lrv  = (float*)carve((size_t)BL_, 4);
  float* lwd  = (float*)carve((size_t)BL_, 4);
  float* wdc  = (float*)carve((size_t)BL_, 4);
  float* wdl  = (float*)carve((size_t)BB_ * NC_, 4);
  float* msm  = (float*)carve((size_t)BB_ * CHUNK_, 4);
  bf16*  Wc   = (bf16*) carve((size_t)BB_ * DIM_ * HID_, 2);
  float* WpK  = (float*)carve((size_t)BB_ * CHUNK_ * DIM_, 4);
  float* kqb  = (float*)carve((size_t)BB_ * CHUNK_ * CHUNK_, 4);
  bf16*  AscT = (bf16*) carve((size_t)BB_ * DIM_ * CHUNK_, 2);
  float* h1   = (float*)carve((size_t)BL_ * HID_, 4);
  float* ha   = (float*)carve((size_t)BL_ * HID_, 4);
  bf16*  gb   = (bf16*) carve((size_t)BL_ * HID_, 2);

  const int T = 256;
  auto conv = [&](const float* src, bf16* dst, long long n) {
    f2bf_kernel<<<dim3((unsigned)((n + T - 1) / T)), T, 0, stream>>>(src, dst, n);
  };

  // ---- one-time (per call) weight bf16 conversion; weights then live in 192MB L2 ----
  conv(Wq,     bWq, (long long)DEPTH_ * HID_ * DIM_);
  conv(Wk,     bWk, (long long)DEPTH_ * HID_ * DIM_);
  conv(Wv,     bWv, (long long)DEPTH_ * DIM_ * DIM_);
  conv(ffn_w1, bW1, (long long)DEPTH_ * HID_ * DIM_);
  conv(ffn_wa, bWa, (long long)DEPTH_ * HID_ * DIM_);
  conv(ffn_wo, bWo, (long long)DEPTH_ * DIM_ * HID_);
  conv(out_w,  bOw, (long long)VOCAB_ * DIM_);

  // ---- embedding ----
  embed_kernel<<<BL_, 256, 0, stream>>>(tokens, emb, x);

  const long long sLH = (long long)LL_ * HID_;

  for (int layer = 0; layer < DEPTH_; layer++) {
    // -------- memory layer --------
    rms_kernel<<<BL_, 256, 0, stream>>>(x, norm_mem + (size_t)layer * DIM_, xn);

    // q, k : [BL, HID] = xn @ Wq/Wk^T   (bf16 out);  v : [BL, DIM] fp32
    gemm_wmma<0, bf16, 4><<<dim3(BL_ / 64, HID_ / 256, 1), 256, 0, stream>>>(
        xn, bWq + (size_t)layer * HID_ * DIM_, qb, BL_, HID_, DIM_, DIM_, DIM_, HID_,
        0, 0, 0, nullptr, nullptr, 0);
    gemm_wmma<0, bf16, 4><<<dim3(BL_ / 64, HID_ / 256, 1), 256, 0, stream>>>(
        xn, bWk + (size_t)layer * HID_ * DIM_, kb, BL_, HID_, DIM_, DIM_, DIM_, HID_,
        0, 0, 0, nullptr, nullptr, 0);
    gemm_wmma<0, float, 4><<<dim3(BL_ / 64, DIM_ / 256, 1), 256, 0, stream>>>(
        xn, bWv + (size_t)layer * DIM_ * DIM_, vf, BL_, DIM_, DIM_, DIM_, DIM_, DIM_,
        0, 0, 0, nullptr, nullptr, 0);

    // kT for the rank-CHUNK W update
    {
      long long n = (long long)BB_ * LL_ * HID_;
      ktrans_kernel<<<dim3((unsigned)((n + T - 1) / T)), T, 0, stream>>>(kb, kT);
    }

    // lr / log_wd GEMVs + per-chunk decay scans
    lrwd_kernel<<<(BL_ + 7) / 8, 256, 0, stream>>>(
        xn, lr_w + (size_t)layer * DIM_, wd_w + (size_t)layer * DIM_,
        lr_b + layer, wd_b + layer, lbl + layer, lbw + layer, lrv, lwd);
    wdscan_kernel<<<1, BB_ * NC_, 0, stream>>>(lwd, wdc, wdl);

    // carry W := W_init[layer] (bf16, both batch slots)
    wcinit_kernel<<<(DIM_ * HID_ + T - 1) / T, T, 0, stream>>>(
        W_init + (size_t)layer * DIM_ * HID_, Wc);

    // -------- chunkwise recurrence --------
    for (int c = 0; c < NC_; c++) {
      int off = c * CHUNK_;
      const bf16* kc = kb + (size_t)off * HID_;
      const bf16* qc = qb + (size_t)off * HID_;

      // kq[b, l, m] = k_l . q_m   (M=N=64, K=2048, batched over b) -- small, JT=2
      gemm_wmma<0, float, 2><<<dim3(1, 1, BB_), 256, 0, stream>>>(
          kc, qc, kqb, CHUNK_, CHUNK_, HID_, HID_, HID_, CHUNK_,
          sLH, sLH, (long long)CHUNK_ * CHUNK_, nullptr, nullptr, 0);

      msum_kernel<<<(BB_ * CHUNK_) / 8, 256, 0, stream>>>(kqb, wdc, msm, off);

      // WpK[b, l, d] = k_l . W_prev[d,:]   (M=64, N=512, K=2048)
      gemm_wmma<0, float, 4><<<dim3(1, DIM_ / 256, BB_), 256, 0, stream>>>(
          kc, Wc, WpK, CHUNK_, DIM_, HID_, HID_, HID_, DIM_,
          sLH, (long long)DIM_ * HID_, (long long)CHUNK_ * DIM_, nullptr, nullptr, 0);

      // y (residual into x) + scaled transposed update operand
      yepi_kernel<<<(BB_ * CHUNK_ * DIM_) / T, T, 0, stream>>>(
          WpK, vf, lrv, wdc, msm, wdl, x, AscT, off, c);

      // W := W * wd_last + AscT @ kT^T    (M=512, N=2048, K=64, in-place bf16)
      gemm_wmma<3, bf16, 4><<<dim3(DIM_ / 64, HID_ / 256, BB_), 256, 0, stream>>>(
          AscT, kT + (size_t)c * HID_ * CHUNK_, Wc, DIM_, HID_, CHUNK_,
          CHUNK_, CHUNK_, HID_,
          (long long)DIM_ * CHUNK_, (long long)NC_ * HID_ * CHUNK_,
          (long long)DIM_ * HID_, nullptr, wdl + c, NC_);
    }

    // -------- FFN --------
    rms_kernel<<<BL_, 256, 0, stream>>>(x, norm_ffn + (size_t)layer * DIM_, xn);
    gemm_wmma<1, float, 4><<<dim3(BL_ / 64, HID_ / 256, 1), 256, 0, stream>>>(
        xn, bW1 + (size_t)layer * HID_ * DIM_, h1, BL_, HID_, DIM_, DIM_, DIM_, HID_,
        0, 0, 0, ffn_b1 + (size_t)layer * HID_, nullptr, 0);
    gemm_wmma<1, float, 4><<<dim3(BL_ / 64, HID_ / 256, 1), 256, 0, stream>>>(
        xn, bWa + (size_t)layer * HID_ * DIM_, ha, BL_, HID_, DIM_, DIM_, DIM_, HID_,
        0, 0, 0, ffn_ba + (size_t)layer * HID_, nullptr, 0);
    {
      long long n = (long long)BL_ * HID_;
      gate_kernel<<<dim3((unsigned)((n + T - 1) / T)), T, 0, stream>>>(h1, ha, gb, n);
    }
    gemm_wmma<2, float, 4><<<dim3(BL_ / 64, DIM_ / 256, 1), 256, 0, stream>>>(
        gb, bWo + (size_t)layer * DIM_ * HID_, x, BL_, DIM_, HID_, HID_, HID_, DIM_,
        0, 0, 0, ffn_bo + (size_t)layer * DIM_, nullptr, 0);
  }

  // -------- final norm + logits --------
  rms_kernel<<<BL_, 256, 0, stream>>>(x, norm_last, xn);
  gemm_wmma<1, float, 4><<<dim3(BL_ / 64, VOCAB_ / 256, 1), 256, 0, stream>>>(
      xn, bOw, logits, BL_, VOCAB_, DIM_, DIM_, DIM_, VOCAB_,
      0, 0, 0, out_b, nullptr, 0);
}
